// LAGEModule_74234214744128
// MI455X (gfx1250) — compile-verified, matrix operand order checked by reading, not run on hardware
//
#include <hip/hip_runtime.h>
#include <hip/hip_bf16.h>

// ---------------------------------------------------------------------------
// LAGE module (GCN branch + multi-head attention branch) for MI455X / gfx1250.
// All matrix contractions use v_wmma_f32_16x16x32_bf16 (wave32).
//  - Flash attention: scores never hit HBM (saves ~3 GB round-trip traffic).
//  - All GEMM/attention B-operands pre-transposed (V is produced transposed by
//    its projection GEMM) so every LDS fragment load is 2 x ds_load_b128.
//  - Bool attention mask packed to bitmasks: 33.5 MB -> 4.2 MB of mask traffic.
// ---------------------------------------------------------------------------

typedef __bf16 bf16;
typedef __attribute__((ext_vector_type(16))) __bf16 v16bf;
typedef __attribute__((ext_vector_type(8)))  float  v8f;

#define B_     8
#define N_     2048
#define DIM_   512
#define H_     8
#define DH_    64
#define LDQK_  1024
#define BN_    (B_ * N_)
#define NEGV   (-1.0e9f)

// ---- WMMA 16x16x32 bf16 fragment helpers (wave32 VGPR layouts, ISA 7.12.2) ----
// 16-bit A (16x32): lane m in [0,15] holds row m, K = {0..7, 16..23};
// lane m+16 holds row m, K = {8..15, 24..31}.
__device__ __forceinline__ int kmap16(int i, int lane) {
  return ((i < 8) ? 0 : 16) + (((i >> 1) & 3) << 1) + (i & 1) + ((lane & 16) ? 8 : 0);
}

__device__ __forceinline__ v8f wmma_bf16(v16bf a, v16bf b, v8f c) {
  // (neg_a, A, neg_b, B, c_mod, C, reuse_a, reuse_b)
  return __builtin_amdgcn_wmma_f32_16x16x32_bf16(false, a, false, b, (short)0, c,
                                                 false, false);
}

// A fragment: element A[row][k] at p[row*ld + k]. Per lane this is two
// contiguous 8-element (16B) runs -> lowers to 2 x ds_load_b128.
__device__ __forceinline__ v16bf frag_A(const bf16* p, int ld) {
  const int lane = threadIdx.x & 31;
  const int row  = lane & 15;
  v16bf a;
#pragma unroll
  for (int i = 0; i < 16; ++i) a[i] = p[row * ld + kmap16(i, lane)];
  return a;
}

// B fragment from TRANSPOSED storage: element B[k][n] at p[n*ld + k].
// Same contiguity as frag_A -> 2 x ds_load_b128.
__device__ __forceinline__ v16bf frag_Bt(const bf16* p, int ld) {
  const int lane = threadIdx.x & 31;
  const int col  = lane & 15;
  v16bf b;
#pragma unroll
  for (int i = 0; i < 16; ++i) b[i] = p[col * ld + kmap16(i, lane)];
  return b;
}

// ---------------------------------------------------------------------------
// f32 -> bf16 cast (grid-stride)
// ---------------------------------------------------------------------------
__global__ void cast_f32_to_bf16(const float* __restrict__ s, bf16* __restrict__ d,
                                 long long n) {
  long long i = (long long)blockIdx.x * blockDim.x + threadIdx.x;
  const long long stride = (long long)gridDim.x * blockDim.x;
  for (; i < n; i += stride) d[i] = (bf16)s[i];
}

// f32 [K x Nn] row-major  ->  bf16 [Nn x K] row-major (B-operand pre-transpose)
__global__ void transpose_cast_bf16(const float* __restrict__ s, bf16* __restrict__ d,
                                    int K, int Nn) {
  long long i = (long long)blockIdx.x * blockDim.x + threadIdx.x;
  const long long total = (long long)K * Nn;
  const long long stride = (long long)gridDim.x * blockDim.x;
  for (; i < total; i += stride) {
    const int n = (int)(i / K);
    const int k = (int)(i % K);
    d[i] = (bf16)s[(long long)k * Nn + n];
  }
}

// Pack bool mask bytes into bitmasks: word w of row i holds keys [32w, 32w+32).
__global__ void pack_mask_bits(const unsigned char* __restrict__ m,
                               unsigned* __restrict__ p, long long nwords) {
  long long i = (long long)blockIdx.x * blockDim.x + threadIdx.x;
  const long long stride = (long long)gridDim.x * blockDim.x;
  for (; i < nwords; i += stride) {
    const uint4* s = reinterpret_cast<const uint4*>(m + i * 32);
    const uint4 q0 = s[0], q1 = s[1];
    unsigned w = 0;
    const unsigned dw[8] = {q0.x, q0.y, q0.z, q0.w, q1.x, q1.y, q1.z, q1.w};
#pragma unroll
    for (int j = 0; j < 8; ++j)
#pragma unroll
      for (int b = 0; b < 4; ++b)
        w |= (((dw[j] >> (8 * b)) & 0xffu) ? 1u : 0u) << (j * 4 + b);
    p[i] = w;
  }
}

// ---------------------------------------------------------------------------
// Tiled WMMA GEMM: C[M,N] = A[M,K] (bf16, row-major) * B[K,N] where B is given
// PRE-TRANSPOSED as BT[N,K] (bf16, row-major).
// Block = 256 threads = 8 waves; block tile 128x128, K-step 32.
// Wave w computes rows [w*16, w*16+16) x all 128 cols: 8 WMMA per A-fragment.
// OMODE: 0 = f32 out (+bias), 1 = bf16 out, 2 = bf16 out TRANSPOSED (C^T).
// Batched via blockIdx.z with element strides sA/sB/sC.
// ---------------------------------------------------------------------------
template <int OMODE>
__global__ void __launch_bounds__(256)
gemm_bt_wmma(const bf16* __restrict__ A, const bf16* __restrict__ BT,
             void* __restrict__ Cout, const float* __restrict__ bias,
             int K, int lda, int ldbt, int ldc,
             long long sA, long long sB, long long sC) {
  __shared__ bf16 As [128][40];  // 32-elem rows padded to 40 (80B = 5 x 16B)
  __shared__ bf16 BsT[128][40];  // B^T tile: 128 cols x 32 k, same padding

  const int tid  = threadIdx.x;
  const int wave = tid >> 5;
  const int lane = tid & 31;
  A  += (long long)blockIdx.z * sA;
  BT += (long long)blockIdx.z * sB;
  const int rowBase = blockIdx.y * 128;
  const int colBase = blockIdx.x * 128;

  v8f acc[8];
#pragma unroll
  for (int n = 0; n < 8; ++n)
#pragma unroll
    for (int i = 0; i < 8; ++i) acc[n][i] = 0.0f;

  for (int k0 = 0; k0 < K; k0 += 32) {
    // Stage A tile (128x32) and BT tile (128x32): 512 16B chunks each,
    // 2 + 2 per thread, fully coalesced 128-bit loads/stores.
#pragma unroll
    for (int t = 0; t < 2; ++t) {
      const int ci = t * 256 + tid;
      const int r  = ci >> 2;          // 4 chunks per 32-elem row
      const int c  = (ci & 3) << 3;
      *reinterpret_cast<uint4*>(&As[r][c]) =
          *reinterpret_cast<const uint4*>(&A[(long long)(rowBase + r) * lda + k0 + c]);
      *reinterpret_cast<uint4*>(&BsT[r][c]) =
          *reinterpret_cast<const uint4*>(&BT[(long long)(colBase + r) * ldbt + k0 + c]);
    }
    // Prefetch next K-tile (global_prefetch_b8) to hide HBM latency on the
    // streamed adj operand.
    if (k0 + 32 < K) {
      const int r = tid >> 1, c = (tid & 1) << 4;
      __builtin_prefetch(&A[(long long)(rowBase + r) * lda + k0 + 32 + c], 0, 0);
      __builtin_prefetch(&BT[(long long)(colBase + r) * ldbt + k0 + 32 + c], 0, 0);
    }
    __syncthreads();

    const v16bf a = frag_A(&As[wave * 16][0], 40);
#pragma unroll
    for (int n = 0; n < 8; ++n) {
      const v16bf b = frag_Bt(&BsT[n * 16][0], 40);
      acc[n] = wmma_bf16(a, b, acc[n]);
    }
    __syncthreads();
  }

  // Epilogue. C layout: lane holds col (lane&15); VGPR r holds row r + 8*(lane>=16).
  const int col0  = lane & 15;
  const int rhalf = (lane & 16) ? 8 : 0;
  bf16*  Cb = (bf16*)Cout + (long long)blockIdx.z * sC;
  float* Cf = (float*)Cout + (long long)blockIdx.z * sC;
#pragma unroll
  for (int n = 0; n < 8; ++n) {
    const int col = colBase + n * 16 + col0;
    const float bv = (OMODE == 0 && bias) ? bias[col] : 0.0f;
#pragma unroll
    for (int r = 0; r < 8; ++r) {
      const long long row = rowBase + wave * 16 + rhalf + r;
      const float v = acc[n][r] + bv;
      if (OMODE == 0)      Cf[row * ldc + col] = v;
      else if (OMODE == 1) Cb[row * ldc + col] = (bf16)v;
      else                 Cb[(long long)col * ldc + row] = (bf16)v;  // C^T
    }
  }
}

// ---------------------------------------------------------------------------
// Flash attention, one (b, h) slice per blockIdx.{z,y}.
// Block = 128 threads = 4 waves, each wave owns 16 query rows; waves share
// LDS-staged 32-key K tiles (row-major) and V tiles (transposed: VT rows are
// key-contiguous, staged with coalesced b128 loads). Every WMMA fragment in
// the loop is 2 x ds_load_b128 -- no 16-bit gathers anywhere.
// Online softmax with shfl_xor row reductions (rows live in the WMMA C-layout:
// lane = col, VGPR index = row). P round-trips through per-wave LDS to
// re-swizzle C-layout -> A-layout. Output written into concat cols [0, 512).
// ---------------------------------------------------------------------------
__global__ void __launch_bounds__(128)
flash_attn_wmma(const bf16* __restrict__ qk, const bf16* __restrict__ vT,
                const unsigned* __restrict__ mbits, bf16* __restrict__ outcat) {
  __shared__ bf16 Ks [32][72];  // [key][dim]   for Q*K^T (frag_Bt on rows)
  __shared__ bf16 VsT[64][40];  // [dim][key]   for P*V   (frag_Bt on rows)
  __shared__ bf16 Ps [4][16][40];

  const int tid  = threadIdx.x;
  const int wave = tid >> 5;
  const int lane = tid & 31;
  const int h = blockIdx.y;
  const int b = blockIdx.z;
  const int qRow0 = blockIdx.x * 64 + wave * 16;
  const long long baseRow = (long long)b * N_;
  const int col0  = lane & 15;
  const int rhalf = (lane & 16) ? 8 : 0;

  // Load and pre-scale Q fragments (dims 0-31 and 32-63), A-layout gather.
  const float scale = 0.125f;  // 64^-0.5
  v16bf aq0, aq1;
  {
    const bf16* qp = qk + (baseRow + qRow0 + (lane & 15)) * LDQK_ + h * DH_;
#pragma unroll
    for (int i = 0; i < 16; ++i) {
      const int d = kmap16(i, lane);
      aq0[i] = (bf16)((float)qp[d]      * scale);
      aq1[i] = (bf16)((float)qp[d + 32] * scale);
    }
  }

  v8f o[4];
#pragma unroll
  for (int n = 0; n < 4; ++n)
#pragma unroll
    for (int i = 0; i < 8; ++i) o[n][i] = 0.0f;
  float m_run[8], l_run[8];
#pragma unroll
  for (int r = 0; r < 8; ++r) { m_run[r] = -1.0e30f; l_run[r] = 0.0f; }

  // Packed mask row base for this batch: 64 u32 words per row.
  const unsigned* mrow = mbits + baseRow * 64;

  for (int j0 = 0; j0 < N_; j0 += 32) {
    __syncthreads();  // previous chunk's LDS reads done before overwrite
    // Stage K chunk [32 keys x 64 dims] from row-major K (coalesced b128).
#pragma unroll
    for (int t = 0; t < 2; ++t) {
      const int ci = t * 128 + tid;
      const int r  = ci >> 3;
      const int c  = (ci & 7) << 3;
      *reinterpret_cast<uint4*>(&Ks[r][c]) = *reinterpret_cast<const uint4*>(
          &qk[(baseRow + j0 + r) * LDQK_ + 512 + h * DH_ + c]);
    }
    // Stage V chunk transposed [64 dims x 32 keys] from V^T (coalesced b128).
#pragma unroll
    for (int t = 0; t < 2; ++t) {
      const int ci = t * 128 + tid;
      const int d  = ci >> 2;          // 4 chunks of 8 keys per dim row
      const int c  = (ci & 3) << 3;
      *reinterpret_cast<uint4*>(&VsT[d][c]) = *reinterpret_cast<const uint4*>(
          &vT[(long long)(h * DH_ + d) * BN_ + baseRow + j0 + c]);
    }
    __syncthreads();

    // S = Q*K^T: two 16x16 tiles (keys 0-15, 16-31). K rows are keys, so the
    // B fragment B[d][n] = K[n][d] is a frag_Bt on the row-major tile.
    v8f s0, s1;
#pragma unroll
    for (int i = 0; i < 8; ++i) { s0[i] = 0.0f; s1[i] = 0.0f; }
    s0 = wmma_bf16(aq0, frag_Bt(&Ks[0][0],   72), s0);
    s0 = wmma_bf16(aq1, frag_Bt(&Ks[0][32],  72), s0);
    s1 = wmma_bf16(aq0, frag_Bt(&Ks[16][0],  72), s1);
    s1 = wmma_bf16(aq1, frag_Bt(&Ks[16][32], 72), s1);

    // Mask (packed bits) + online softmax.
    const int w = j0 >> 5;
    float corr[8];
#pragma unroll
    for (int r = 0; r < 8; ++r) {
      const long long irow = qRow0 + rhalf + r;
      const unsigned bits = mrow[irow * 64 + w];  // uniform across 16 lanes
      const float x0 = ((bits >> col0)        & 1u) ? NEGV : s0[r];
      const float x1 = ((bits >> (col0 + 16)) & 1u) ? NEGV : s1[r];
      float mx = fmaxf(x0, x1);
#pragma unroll
      for (int off = 1; off < 16; off <<= 1) mx = fmaxf(mx, __shfl_xor(mx, off, 32));
      const float mnew = fmaxf(m_run[r], mx);
      const float c  = __expf(m_run[r] - mnew);
      const float p0 = __expf(x0 - mnew);
      const float p1 = __expf(x1 - mnew);
      float ps = p0 + p1;
#pragma unroll
      for (int off = 1; off < 16; off <<= 1) ps += __shfl_xor(ps, off, 32);
      l_run[r] = l_run[r] * c + ps;
      m_run[r] = mnew;
      corr[r]  = c;
      // Stash P in LDS (C-layout positions) for the A-layout re-gather.
      Ps[wave][rhalf + r][col0]      = (bf16)p0;
      Ps[wave][rhalf + r][col0 + 16] = (bf16)p1;
    }

    // Rescale running output, then O += P * V (4 dim-tiles of 16).
#pragma unroll
    for (int n = 0; n < 4; ++n)
#pragma unroll
      for (int r = 0; r < 8; ++r) o[n][r] *= corr[r];

    const v16bf ap = frag_A(&Ps[wave][0][0], 40);  // 16 rows x 32 keys
#pragma unroll
    for (int n = 0; n < 4; ++n) {
      const v16bf bv = frag_Bt(&VsT[n * 16][0], 40);  // B[key][dim] via V^T
      o[n] = wmma_bf16(ap, bv, o[n]);
    }
  }

  // Normalize and write into concat buffer columns [h*64, h*64+64).
#pragma unroll
  for (int r = 0; r < 8; ++r) {
    const long long irow = baseRow + qRow0 + rhalf + r;
    const float inv = 1.0f / l_run[r];
#pragma unroll
    for (int n = 0; n < 4; ++n)
      outcat[irow * (2 * DIM_) + h * DH_ + n * 16 + col0] = (bf16)(o[n][r] * inv);
  }
}

// ---------------------------------------------------------------------------
// Launcher
// ---------------------------------------------------------------------------
extern "C" void kernel_launch(void* const* d_in, const int* in_sizes, int n_in,
                              void* d_out, int out_size, void* d_ws, size_t ws_size,
                              hipStream_t stream) {
  (void)in_sizes; (void)n_in; (void)out_size; (void)ws_size;
  const float*         x        = (const float*)d_in[0];
  const float*         adj      = (const float*)d_in[1];
  const unsigned char* att_mask = (const unsigned char*)d_in[2];
  const float*         W_gcn    = (const float*)d_in[3];
  const float*         W_qkv    = (const float*)d_in[4];
  const float*         W_out    = (const float*)d_in[5];
  const float*         b_out    = (const float*)d_in[6];
  float*               out      = (float*)d_out;

  // Carve staging buffers out of the workspace (256B-aligned).
  char*  w   = (char*)d_ws;
  size_t off = 0;
  auto alloc = [&](size_t bytes) -> void* {
    void* p = (void*)(w + off);
    off += (bytes + 255) & ~(size_t)255;
    return p;
  };
  bf16* x_bf   = (bf16*)alloc(sizeof(bf16) * BN_ * DIM_);
  bf16* adj_bf = (bf16*)alloc(sizeof(bf16) * B_ * N_ * N_);
  bf16* wgT    = (bf16*)alloc(sizeof(bf16) * DIM_ * DIM_);          // W_gcn^T
  bf16* wqT    = (bf16*)alloc(sizeof(bf16) * DIM_ * 3 * DIM_);      // W_qkv^T
  bf16* woT    = (bf16*)alloc(sizeof(bf16) * 2 * DIM_ * DIM_);      // W_out^T
  bf16* xgT    = (bf16*)alloc(sizeof(bf16) * BN_ * DIM_);           // (x W_gcn)^T
  bf16* qk_bf  = (bf16*)alloc(sizeof(bf16) * BN_ * 2 * DIM_);       // [BN, 1024]
  bf16* vT     = (bf16*)alloc(sizeof(bf16) * DIM_ * BN_);           // V^T [512, BN]
  bf16* cat_bf = (bf16*)alloc(sizeof(bf16) * BN_ * 2 * DIM_);
  unsigned* mpk = (unsigned*)alloc(sizeof(unsigned) * (size_t)B_ * N_ * (N_ / 32));

  // 1) Precision casts + B-operand pre-transposes + mask bit-packing.
  cast_f32_to_bf16<<<4096, 256, 0, stream>>>(x,   x_bf,   (long long)BN_ * DIM_);
  cast_f32_to_bf16<<<4096, 256, 0, stream>>>(adj, adj_bf, (long long)B_ * N_ * N_);
  transpose_cast_bf16<<<1024, 256, 0, stream>>>(W_gcn, wgT, DIM_, DIM_);
  transpose_cast_bf16<<<3072, 256, 0, stream>>>(W_qkv, wqT, DIM_, 3 * DIM_);
  transpose_cast_bf16<<<2048, 256, 0, stream>>>(W_out, woT, 2 * DIM_, DIM_);
  pack_mask_bits<<<4096, 256, 0, stream>>>(att_mask, mpk,
                                           (long long)B_ * N_ * (N_ / 32));

  const dim3 blk(256);
  // 2) xg^T = (x @ W_gcn)^T    [16384 x 512, K=512] -> bf16 transposed
  gemm_bt_wmma<2><<<dim3(DIM_ / 128, BN_ / 128, 1), blk, 0, stream>>>(
      x_bf, wgT, xgT, nullptr, DIM_, DIM_, DIM_, BN_, 0, 0, 0);
  // 3a) qk = x @ W_qkv[:, 0:1024]      [16384 x 1024, K=512] -> bf16
  gemm_bt_wmma<1><<<dim3(2 * DIM_ / 128, BN_ / 128, 1), blk, 0, stream>>>(
      x_bf, wqT, qk_bf, nullptr, DIM_, DIM_, DIM_, 2 * DIM_, 0, 0, 0);
  // 3b) v^T = (x @ W_qkv[:, 1024:1536])^T  [512 x 16384] -> bf16 transposed
  gemm_bt_wmma<2><<<dim3(DIM_ / 128, BN_ / 128, 1), blk, 0, stream>>>(
      x_bf, wqT + (size_t)2 * DIM_ * DIM_, vT, nullptr, DIM_, DIM_, DIM_, BN_,
      0, 0, 0);
  // 4) gcn branch: concat[:, 512:1024] = adj @ xg, batched over B.
  //    BT = xg^T [512 x 16384]; batch b uses columns [b*2048, b*2048+2048).
  gemm_bt_wmma<1><<<dim3(DIM_ / 128, N_ / 128, B_), blk, 0, stream>>>(
      adj_bf, xgT, (bf16*)cat_bf + DIM_, nullptr, N_, N_, BN_, 2 * DIM_,
      (long long)N_ * N_, (long long)N_, (long long)N_ * 2 * DIM_);
  // 5) attention branch: concat[:, 0:512] (flash, scores stay on-chip).
  flash_attn_wmma<<<dim3(N_ / 64, H_, B_), dim3(128), 0, stream>>>(
      qk_bf, vT, mpk, cat_bf);
  // 6) out = concat @ W_out + b_out   [16384 x 512, K=1024] -> f32
  gemm_bt_wmma<0><<<dim3(DIM_ / 128, BN_ / 128, 1), blk, 0, stream>>>(
      cat_bf, woT, out, b_out, 2 * DIM_, 2 * DIM_, 2 * DIM_, DIM_, 0, 0, 0);
}
